// TransformerBlock_3702261809539
// MI455X (gfx1250) — matrix operand order repaired and checked by reading, not compiled, not run on hardware
//
#include <hip/hip_runtime.h>
#include <hip/hip_bf16.h>
#include <math.h>

// ---------------------------------------------------------------------------
// Types / helpers
// ---------------------------------------------------------------------------
typedef __bf16 bf16;
typedef __attribute__((ext_vector_type(16))) __bf16 v16bf;
typedef __attribute__((ext_vector_type(8)))  float  v8f;

#define DEV static __device__ __forceinline__

union FragU { uint4 q[2]; v16bf v; };

// Low 32 bits of a generic pointer into LDS == LDS byte address.
DEV uint32_t lds_addr(const void* p) { return (uint32_t)(uintptr_t)p; }

// Async global->LDS copy, 16B per lane, GVS addressing (saddr + 32-bit voffset).
// Tracked by ASYNCcnt (CDNA5 ISA).
DEV void async_b128(uint32_t lds, uint32_t voff, const void* sbase) {
  asm volatile("global_load_async_to_lds_b128 %0, %1, %2"
               :: "v"(lds), "v"(voff), "s"(sbase) : "memory");
}
DEV void wait_async_0() { asm volatile("s_wait_asynccnt 0" ::: "memory"); }
DEV void wait_async_4() { asm volatile("s_wait_asynccnt 4" ::: "memory"); }
DEV void wait_async_6() { asm volatile("s_wait_asynccnt 6" ::: "memory"); }

DEV v8f wmma_bf16(v16bf a, v16bf b, v8f c) {
  // D(16x16,f32) = A(16x32,bf16) * B(32x16,bf16) + C
  return __builtin_amdgcn_wmma_f32_16x16x32_bf16(
      false, a, false, b, (short)0, c, false, false);
}

// A fragment: 16x32 bf16 tile at (mbase,kbase); lanes 0-15 hold K {0..7,16..23},
// lanes 16-31 hold K {8..15,24..31}. Two aligned 16B LDS loads per lane.
DEV v16bf frag_a(const bf16* p, int ld, int mbase, int kbase, int lane) {
  const int m  = mbase + (lane & 15);
  const int kh = (lane < 16) ? 0 : 8;
  const bf16* row = p + m * ld + kbase;
  FragU u;
  u.q[0] = *(const uint4*)(row + kh);
  u.q[1] = *(const uint4*)(row + 16 + kh);
  return u.v;
}

// B fragment from an N x K row-major tile (A * W^T). Lane holds column N=L%16;
// lanes 0-15 K {0..7,16..23}, lanes 16-31 K {8..15,24..31}. Two aligned 16B loads.
DEV v16bf frag_b(const bf16* p, int ld, int nbase, int kbase, int lane) {
  const int n  = nbase + (lane & 15);
  const int k0 = kbase + ((lane < 16) ? 0 : 8);
  const bf16* row = p + n * ld + k0;
  FragU u;
  u.q[0] = *(const uint4*)(row);
  u.q[1] = *(const uint4*)(row + 16);
  return u.v;
}

// ---------------------------------------------------------------------------
// Weight conversion kernels
// ---------------------------------------------------------------------------
__global__ void cvt_f32_bf16(const float* __restrict__ in, bf16* __restrict__ out, int n) {
  const int i = blockIdx.x * 256 + threadIdx.x;
  if (i < n) out[i] = (bf16)in[i];
}

// W_O [H, D, Dh] -> Wo_b [D][(H,Dh)] so out-proj is a plain A*W^T GEMM.
__global__ void cvt_wo_bf16(const float* __restrict__ WO, bf16* __restrict__ out) {
  const int i = blockIdx.x * 256 + threadIdx.x;   // i < 1024*1024
  const int d = i >> 10, rem = i & 1023, nh = rem >> 6, hh = rem & 63;
  out[i] = (bf16)WO[((long)nh * 1024 + d) * 64 + hh];
}

// ---------------------------------------------------------------------------
// LayerNorm (fp32 in, bf16 out), one 1024-wide row per 256-thread block
// ---------------------------------------------------------------------------
__global__ __launch_bounds__(256)
void layernorm_to_bf16(const float* __restrict__ x, const float* __restrict__ g,
                       const float* __restrict__ bta, bf16* __restrict__ out) {
  const int row = blockIdx.x;
  const int tid = threadIdx.x;
  const float* xr = x + (long)row * 1024;
  __shared__ float red[256];

  float s = 0.f;
#pragma unroll
  for (int i = 0; i < 4; ++i) s += xr[tid + i * 256];
  red[tid] = s; __syncthreads();
  for (int o = 128; o > 0; o >>= 1) {
    if (tid < o) red[tid] += red[tid + o];
    __syncthreads();
  }
  const float mu = red[0] * (1.0f / 1024.0f);
  __syncthreads();

  float v = 0.f;
#pragma unroll
  for (int i = 0; i < 4; ++i) { const float d = xr[tid + i * 256] - mu; v += d * d; }
  red[tid] = v; __syncthreads();
  for (int o = 128; o > 0; o >>= 1) {
    if (tid < o) red[tid] += red[tid + o];
    __syncthreads();
  }
  const float inv = rsqrtf(red[0] * (1.0f / 1024.0f) + 1e-5f);

#pragma unroll
  for (int i = 0; i < 4; ++i) {
    const int c = tid + i * 256;
    out[(long)row * 1024 + c] = (bf16)((xr[c] - mu) * inv * g[c] + bta[c]);
  }
}

// ---------------------------------------------------------------------------
// WMMA bf16 GEMM: C[M,N] = A[M,K] * W[N,K]^T + bias.
// 256 threads (8 waves, 4x2), block tile 128x64, K-stage 64, double-buffered
// LDS fed by async global->LDS copies pipelined on ASYNCcnt.
// ---------------------------------------------------------------------------
enum { EPI_QKV = 0, EPI_GELU_BF16 = 1, EPI_BIAS_RES_F32 = 2, EPI_V_T = 3 };

template <int EPI>
__global__ __launch_bounds__(256)
void gemm_bf16_wmma(const bf16* __restrict__ A, const bf16* __restrict__ W,
                    const float* __restrict__ bias, void* __restrict__ out,
                    const float* __restrict__ resid, int M, int N, int K) {
  __shared__ alignas(16) bf16 As[2][128][64];   // 32 KB
  __shared__ alignas(16) bf16 Bs[2][64][64];    // 16 KB

  const int tid  = threadIdx.x;
  const int lane = tid & 31;
  const int wave = tid >> 5;
  const int wm   = wave >> 1;               // 0..3
  const int wn   = wave & 1;                // 0..1
  const long bm  = (long)blockIdx.y * 128;
  const long bn  = (long)blockIdx.x * 64;

  v8f acc[2][2] = {};

  // per-stage async issue: A 16KB (4 x b128/thread) + B 8KB (2 x b128/thread)
  auto issue_stage = [&](int s) {
    const int buf = s & 1;
    const bf16* Abase = A + bm * K + (long)s * 64;
    const bf16* Wbase = W + bn * K + (long)s * 64;
#pragma unroll
    for (int i = 0; i < 4; ++i) {
      const int slot = tid + i * 256;
      const int r = slot >> 3, sg = (slot & 7) * 8;
      async_b128(lds_addr(&As[buf][r][sg]), (uint32_t)(((long)r * K + sg) * 2), Abase);
    }
#pragma unroll
    for (int i = 0; i < 2; ++i) {
      const int slot = tid + i * 256;
      const int r = slot >> 3, sg = (slot & 7) * 8;
      async_b128(lds_addr(&Bs[buf][r][sg]), (uint32_t)(((long)r * K + sg) * 2), Wbase);
    }
  };

  const int nstage = K >> 6;
  issue_stage(0);
  for (int s = 0; s < nstage; ++s) {
    if (s + 1 < nstage) {
      issue_stage(s + 1);
      wait_async_6();   // in-order completion => stage s has landed
    } else {
      wait_async_0();
    }
    __syncthreads();

    const bf16* a  = &As[s & 1][0][0];
    const bf16* bw = &Bs[s & 1][0][0];
#pragma unroll
    for (int kk = 0; kk < 64; kk += 32) {
      const v16bf a0 = frag_a(a,  64, wm * 32,      kk, lane);
      const v16bf a1 = frag_a(a,  64, wm * 32 + 16, kk, lane);
      const v16bf b0 = frag_b(bw, 64, wn * 32,      kk, lane);
      const v16bf b1 = frag_b(bw, 64, wn * 32 + 16, kk, lane);
      acc[0][0] = wmma_bf16(a0, b0, acc[0][0]);
      acc[0][1] = wmma_bf16(a0, b1, acc[0][1]);
      acc[1][0] = wmma_bf16(a1, b0, acc[1][0]);
      acc[1][1] = wmma_bf16(a1, b1, acc[1][1]);
    }
    __syncthreads();
  }

  // Epilogue. C layout: VGPR r -> M = r (lanes 0-15) / r+8 (lanes 16-31), N = lane%16.
#pragma unroll
  for (int i = 0; i < 2; ++i)
#pragma unroll
    for (int j = 0; j < 2; ++j)
#pragma unroll
      for (int r = 0; r < 8; ++r) {
        const long m = bm + wm * 32 + i * 16 + ((lane < 16) ? r : r + 8);
        const long n = bn + wn * 32 + j * 16 + (lane & 15);
        const float v = acc[i][j][r] + bias[n];
        if (EPI == EPI_QKV) {
          // m = b*2048 + t ; n = head*64 + hh ; store [B,H,S,Dh] bf16
          const long bb = m >> 11, t = m & 2047;
          const long hd = n >> 6,  hh = n & 63;
          ((bf16*)out)[(((bb * 16 + hd) * 2048) + t) * 64 + hh] = (bf16)v;
        } else if (EPI == EPI_V_T) {
          // store V transposed: [B,H,Dh,S] so attention reads it as N x K
          const long bb = m >> 11, t = m & 2047;
          const long hd = n >> 6,  hh = n & 63;
          ((bf16*)out)[(((bb * 16 + hd) * 64) + hh) * 2048 + t] = (bf16)v;
        } else if (EPI == EPI_GELU_BF16) {
          const float g = 0.5f * v * (1.0f + erff(v * 0.70710678118654752f));
          ((bf16*)out)[m * (long)N + n] = (bf16)g;
        } else {  // EPI_BIAS_RES_F32
          ((float*)out)[m * (long)N + n] = v + resid[m * (long)N + n];
        }
      }
}

// ---------------------------------------------------------------------------
// Flash attention: one (b, head, 64-query tile) per 256-thread block.
// K/V chunks double-buffered via async->LDS; scores & P*V on WMMA;
// online softmax parallelized 4 threads/row.
// V is pre-transposed in global memory: Vt[B,H,Dh,S].
// ---------------------------------------------------------------------------
__global__ __launch_bounds__(256)
void attn_flash_wmma(const bf16* __restrict__ Q, const bf16* __restrict__ Kt,
                     const bf16* __restrict__ Vt, bf16* __restrict__ ctx) {
  __shared__ alignas(16) bf16  Qs[64][64];        // 8 KB
  __shared__ alignas(16) bf16  Ks[2][64][64];     // 16 KB  [key][dh]
  __shared__ alignas(16) bf16  Vs[2][64][64];     // 16 KB  [dh][key]
  __shared__ alignas(16) float Ss[64][64];        // 16 KB
  __shared__ alignas(16) bf16  Ps[64][64];        // 8 KB
  __shared__ float mrun[64], lrun[64], alph[64];
  __shared__ float pmax[64][4], psum[64][4];

  const int tid  = threadIdx.x, lane = tid & 31, wave = tid >> 5;
  const int wm   = wave >> 1, wn = wave & 1;
  const int qt   = blockIdx.x, h = blockIdx.y, b = blockIdx.z;
  const long bh  = (long)b * 16 + h;
  const bf16* Qg = Q  + (bh * 2048 + (long)qt * 64) * 64;
  const bf16* Vg = Vt + (bh * 64) * 2048;    // rows = dh, stride S

  if (tid < 64) { mrun[tid] = -1e30f; lrun[tid] = 0.0f; }

  // Q tile: contiguous 8 KB, 2 x b128 per thread
#pragma unroll
  for (int i = 0; i < 2; ++i) {
    const int slot = tid + i * 256;
    async_b128(lds_addr(&Qs[0][0] + slot * 8), (uint32_t)(slot * 16), Qg);
  }

  auto issue_kv = [&](int c) {
    const int buf = c & 1;
    const bf16* Kg = Kt + (bh * 2048 + (long)c * 64) * 64;  // contiguous 8 KB
#pragma unroll
    for (int i = 0; i < 2; ++i) {
      const int slot = tid + i * 256;
      async_b128(lds_addr(&Ks[buf][0][0] + slot * 8), (uint32_t)(slot * 16), Kg);
      const int dh = slot >> 3, sg = (slot & 7) * 8;        // V rows strided by S
      async_b128(lds_addr(&Vs[buf][dh][sg]),
                 (uint32_t)((((long)dh * 2048) + c * 64 + sg) * 2), Vg);
    }
  };

  v8f oacc[2] = {};
  issue_kv(0);

  for (int c = 0; c < 32; ++c) {
    if (c + 1 < 32) {
      issue_kv(c + 1);
      wait_async_4();   // in-order => Q + chunk c have landed
    } else {
      wait_async_0();
    }
    __syncthreads();
    const bf16* ks = &Ks[c & 1][0][0];
    const bf16* vs = &Vs[c & 1][0][0];

    // S = Q * K^T  (64x64, keys are B-matrix rows)
    v8f sacc[2] = {};
#pragma unroll
    for (int kk = 0; kk < 64; kk += 32) {
      const v16bf aq = frag_a(&Qs[0][0], 64, wm * 16, kk, lane);
#pragma unroll
      for (int j = 0; j < 2; ++j) {
        const v16bf bk = frag_b(ks, 64, wn * 32 + j * 16, kk, lane);
        sacc[j] = wmma_bf16(aq, bk, sacc[j]);
      }
    }
#pragma unroll
    for (int j = 0; j < 2; ++j)
#pragma unroll
      for (int r = 0; r < 8; ++r) {
        const int m = wm * 16 + ((lane < 16) ? r : r + 8);
        const int n = wn * 32 + j * 16 + (lane & 15);
        Ss[m][n] = sacc[j][r] * 0.125f;          // 1/sqrt(Dh)
      }
    __syncthreads();

    // Online softmax: 4 threads per row, 16 columns each
    {
      const int row = tid >> 2, qq = (tid & 3) * 16;
      float s16[16];
#pragma unroll
      for (int i = 0; i < 4; ++i) {
        const float4 v4 = *(const float4*)&Ss[row][qq + i * 4];
        s16[i * 4 + 0] = v4.x; s16[i * 4 + 1] = v4.y;
        s16[i * 4 + 2] = v4.z; s16[i * 4 + 3] = v4.w;
      }
      float mloc = -1e30f;
#pragma unroll
      for (int i = 0; i < 16; ++i) mloc = fmaxf(mloc, s16[i]);
      pmax[row][tid & 3] = mloc;
      __syncthreads();
      const float mold = mrun[row];
      const float mnew = fmaxf(mold,
          fmaxf(fmaxf(pmax[row][0], pmax[row][1]),
                fmaxf(pmax[row][2], pmax[row][3])));
      union { bf16 hh[16]; uint4 q[2]; } pu;
      float sloc = 0.f;
#pragma unroll
      for (int i = 0; i < 16; ++i) {
        const float p = __expf(s16[i] - mnew);
        pu.hh[i] = (bf16)p;
        sloc += p;
      }
      *(uint4*)&Ps[row][qq]     = pu.q[0];
      *(uint4*)&Ps[row][qq + 8] = pu.q[1];
      psum[row][tid & 3] = sloc;
      __syncthreads();
      if ((tid & 3) == 0) {
        const float al = __expf(mold - mnew);
        lrun[row] = lrun[row] * al +
                    psum[row][0] + psum[row][1] + psum[row][2] + psum[row][3];
        mrun[row] = mnew;
        alph[row] = al;
      }
    }
    __syncthreads();

    // Rescale running O, then O += P * V   (V already [dh][key] in LDS)
#pragma unroll
    for (int j = 0; j < 2; ++j)
#pragma unroll
      for (int r = 0; r < 8; ++r) {
        const int m = wm * 16 + ((lane < 16) ? r : r + 8);
        oacc[j][r] *= alph[m];
      }
#pragma unroll
    for (int kk = 0; kk < 64; kk += 32) {
      const v16bf ap = frag_a(&Ps[0][0], 64, wm * 16, kk, lane);
#pragma unroll
      for (int j = 0; j < 2; ++j) {
        const v16bf bv = frag_b(vs, 64, wn * 32 + j * 16, kk, lane);
        oacc[j] = wmma_bf16(ap, bv, oacc[j]);
      }
    }
    __syncthreads();
  }

  // ctx[b, t, head*64 + dh] (bf16), normalized by running sum
#pragma unroll
  for (int j = 0; j < 2; ++j)
#pragma unroll
    for (int r = 0; r < 8; ++r) {
      const int  m  = wm * 16 + ((lane < 16) ? r : r + 8);
      const int  dh = wn * 32 + j * 16 + (lane & 15);
      const long t  = (long)qt * 64 + m;
      ctx[((long)b * 2048 + t) * 1024 + h * 64 + dh] = (bf16)(oacc[j][r] / lrun[m]);
    }
}

// ---------------------------------------------------------------------------
// Host-side orchestration
// ---------------------------------------------------------------------------
extern "C" void kernel_launch(void* const* d_in, const int* in_sizes, int n_in,
                              void* d_out, int out_size, void* d_ws, size_t ws_size,
                              hipStream_t stream) {
  (void)in_sizes; (void)n_in; (void)out_size; (void)ws_size;
  const float* x    = (const float*)d_in[0];
  const float* ln1w = (const float*)d_in[1];
  const float* ln1b = (const float*)d_in[2];
  const float* WQ   = (const float*)d_in[3];
  const float* bQ   = (const float*)d_in[4];
  const float* WK   = (const float*)d_in[5];
  const float* bK   = (const float*)d_in[6];
  const float* WV   = (const float*)d_in[7];
  const float* bV   = (const float*)d_in[8];
  const float* WO   = (const float*)d_in[9];
  const float* bO   = (const float*)d_in[10];
  const float* ln2w = (const float*)d_in[11];
  const float* ln2b = (const float*)d_in[12];
  const float* Win  = (const float*)d_in[13];
  const float* bin  = (const float*)d_in[14];
  const float* Wout = (const float*)d_in[15];
  const float* bout = (const float*)d_in[16];
  float* outp = (float*)d_out;

  const long M = 8192;  // B*S
  char* ws = (char*)d_ws;
  size_t off = 0;
  auto alloc = [&](size_t bytes) {
    char* p = ws + off;
    off += (bytes + 255) & ~(size_t)255;
    return p;
  };
  bf16* xln    = (bf16*)alloc(M * 1024 * 2);
  bf16* wq_b   = (bf16*)alloc((size_t)1024 * 1024 * 2);
  bf16* wk_b   = (bf16*)alloc((size_t)1024 * 1024 * 2);
  bf16* wv_b   = (bf16*)alloc((size_t)1024 * 1024 * 2);
  bf16* wo_b   = (bf16*)alloc((size_t)1024 * 1024 * 2);
  bf16* win_b  = (bf16*)alloc((size_t)4096 * 1024 * 2);
  bf16* wout_b = (bf16*)alloc((size_t)1024 * 4096 * 2);
  bf16* Qb     = (bf16*)alloc(M * 64 * 16 * 2);
  bf16* Kb     = (bf16*)alloc(M * 64 * 16 * 2);
  bf16* Vb     = (bf16*)alloc(M * 64 * 16 * 2);   // [B,H,Dh,S]
  bf16* ctxb   = (bf16*)alloc(M * 1024 * 2);
  float* xattn = (float*)alloc(M * 1024 * 4);
  bf16* xln2   = (bf16*)alloc(M * 1024 * 2);
  bf16* acts   = (bf16*)alloc(M * 4096 * 2);

  const dim3 thr(256);

  // Weights -> bf16 (deterministic, every call)
  cvt_f32_bf16<<<4096,  thr, 0, stream>>>(WQ,   wq_b,   1048576);
  cvt_f32_bf16<<<4096,  thr, 0, stream>>>(WK,   wk_b,   1048576);
  cvt_f32_bf16<<<4096,  thr, 0, stream>>>(WV,   wv_b,   1048576);
  cvt_wo_bf16 <<<4096,  thr, 0, stream>>>(WO,   wo_b);
  cvt_f32_bf16<<<16384, thr, 0, stream>>>(Win,  win_b,  4194304);
  cvt_f32_bf16<<<16384, thr, 0, stream>>>(Wout, wout_b, 4194304);

  // LN1
  layernorm_to_bf16<<<8192, thr, 0, stream>>>(x, ln1w, ln1b, xln);

  // QKV projections (N=1024, K=1024); V stored transposed [B,H,Dh,S]
  gemm_bf16_wmma<EPI_QKV><<<dim3(16, 64), thr, 0, stream>>>(
      xln, wq_b, bQ, Qb, nullptr, 8192, 1024, 1024);
  gemm_bf16_wmma<EPI_QKV><<<dim3(16, 64), thr, 0, stream>>>(
      xln, wk_b, bK, Kb, nullptr, 8192, 1024, 1024);
  gemm_bf16_wmma<EPI_V_T><<<dim3(16, 64), thr, 0, stream>>>(
      xln, wv_b, bV, Vb, nullptr, 8192, 1024, 1024);

  // Attention (flash, fused)
  attn_flash_wmma<<<dim3(32, 16, 4), thr, 0, stream>>>(Qb, Kb, Vb, ctxb);

  // Output projection + residual(x) -> x_attn (fp32)
  gemm_bf16_wmma<EPI_BIAS_RES_F32><<<dim3(16, 64), thr, 0, stream>>>(
      ctxb, wo_b, bO, xattn, x, 8192, 1024, 1024);

  // LN2
  layernorm_to_bf16<<<8192, thr, 0, stream>>>(xattn, ln2w, ln2b, xln2);

  // MLP in + exact GELU (N=4096)
  gemm_bf16_wmma<EPI_GELU_BF16><<<dim3(64, 64), thr, 0, stream>>>(
      xln2, win_b, bin, acts, nullptr, 8192, 4096, 1024);

  // MLP out + residual(x_attn) -> d_out (fp32), K=4096
  gemm_bf16_wmma<EPI_BIAS_RES_F32><<<dim3(16, 64), thr, 0, stream>>>(
      acts, wout_b, bout, outp, xattn, 8192, 1024, 4096);
}
